// ProprioceptionEncoder_65987877535816
// MI455X (gfx1250) — matrix operand order repaired and checked by reading, not compiled, not run on hardware
//
#include <hip/hip_runtime.h>
#include <hip/hip_bf16.h>

typedef _Float16 half8  __attribute__((ext_vector_type(8)));
typedef _Float16 v16h   __attribute__((ext_vector_type(16)));
typedef float    v8f    __attribute__((ext_vector_type(8)));

#define ROWS   64    // rows per block (four 16-row M-tiles)
#define BLK    512   // threads per block (16 waves)
#define FSTR   392   // 384 + 8 pad (f16), keeps 16B alignment
#define HSTR   520   // 512 + 8 pad (f16)

// ---- group index maps into the per-row encoded vector enc[70] ----
// enc layout: p(50) | t(4) | ee(12) | internal(4)
__device__ const int FG[8]  = {42,43,44,45, 50,51,52,53};
__device__ const int WG[18] = {8,9,10,11,12,13, 22,23,24,25,26,27, 36,37,38,39,40,41};
__device__ const int AG[24] = {0,1,2,3,4,5,6,7, 14,15,16,17,18,19,20,21, 28,29,30,31,32,33,34,35};
__device__ const int HG[4]  = {46,47,48,49};
__device__ const int EG[12] = {54,55,56,57,58,59,60,61,62,63,64,65};
__device__ const int IG[4]  = {66,67,68,69};

// ---------------- weight pre-pack: KxN fp32 row-major -> WMMA B fragments ---------------
// Fragment f = kb*(N/16) + nt is a 32x16 (KxN) block, stored lane-major:
//   out[f*512 + lane*16 + i] = W[(kb*32 + (lane<16?0:8) + (i<8 ? i : 16 + (i-8))) * N + nt*16 + (lane&15)]
__global__ void pack_wmma_b(const float* __restrict__ W, _Float16* __restrict__ out,
                            int N, int total) {
    int idx = blockIdx.x * 256 + threadIdx.x;
    if (idx >= total) return;
    int f      = idx >> 9;
    int within = idx & 511;
    int l      = within >> 4;
    int i      = within & 15;
    int ntiles = N >> 4;
    int kb     = f / ntiles;
    int nt     = f - kb * ntiles;
    int k      = kb * 32 + ((l < 16) ? 0 : 8) + ((i < 8) ? i : 16 + (i - 8));
    int col    = nt * 16 + (l & 15);
    out[(size_t)f * 512 + within] = (_Float16)W[(size_t)k * N + col];
}

// ---------------- small-MLP helpers (fp32 VALU) ----------------
template <int K, int N>
__device__ inline void layer1(const float* __restrict__ encRow, const int* __restrict__ map,
                              const float* __restrict__ W, const float* __restrict__ bv,
                              float* __restrict__ out, int c, int step) {
    for (int n = c; n < N; n += step) {
        float acc = bv[n];
#pragma unroll
        for (int k = 0; k < K; ++k) acc = fmaf(encRow[map[k]], W[k * N + n], acc);
        out[n] = fmaxf(acc, 0.0f);
    }
}

template <int K, int N>
__device__ inline void layer2(const float* __restrict__ in,
                              const float* __restrict__ W, const float* __restrict__ bv,
                              _Float16* __restrict__ out, int c, int step) {
    for (int n = c; n < N; n += step) {
        float acc = bv[n];
#pragma unroll
        for (int k = 0; k < K; ++k) acc = fmaf(in[k], W[k * N + n], acc);
        out[n] = (_Float16)fmaxf(acc, 0.0f);
    }
}

// phase-1 scratch overlays GEMM2's hidden buffer (disjoint lifetimes)
struct Phase1 {
    float enc[ROWS][72];     // 70 encoded features / row
    float h1s[ROWS][228];    // small-MLP hidden (224 used)
};
union Overlay {
    Phase1   p1;
    _Float16 hbuf[ROWS][HSTR];   // hidden 64x512 f16
};

// ---------------- fused forward kernel: 64 rows per 512-thread block ----------------
__global__ void __launch_bounds__(BLK)
proprio_fused_kernel(const float* __restrict__ proprio, const float* __restrict__ tactile,
                     const float* __restrict__ ee, const float* __restrict__ internal_,
                     const float* __restrict__ f_w1, const float* __restrict__ f_b1,
                     const float* __restrict__ f_w2, const float* __restrict__ f_b2,
                     const float* __restrict__ w_w1, const float* __restrict__ w_b1,
                     const float* __restrict__ w_w2, const float* __restrict__ w_b2,
                     const float* __restrict__ a_w1, const float* __restrict__ a_b1,
                     const float* __restrict__ a_w2, const float* __restrict__ a_b2,
                     const float* __restrict__ h_w1, const float* __restrict__ h_b1,
                     const float* __restrict__ h_w2, const float* __restrict__ h_b2,
                     const float* __restrict__ e_w1, const float* __restrict__ e_b1,
                     const float* __restrict__ e_w2, const float* __restrict__ e_b2,
                     const float* __restrict__ i_w1, const float* __restrict__ i_b1,
                     const float* __restrict__ i_w2, const float* __restrict__ i_b2,
                     const _Float16* __restrict__ pW1, const float* __restrict__ fu_b1,
                     const _Float16* __restrict__ pW2, const float* __restrict__ fu_b2,
                     const float* __restrict__ ln_g, const float* __restrict__ ln_b,
                     float* __restrict__ out) {
    __shared__ Overlay  ovl;
    __shared__ _Float16 fusedS[ROWS][FSTR];   // fused 64x384 f16
    __shared__ float    sums[ROWS], sums2[ROWS];

    const int tid  = threadIdx.x;
    const int r    = tid & 63;        // row within tile (0..63)
    const int c    = tid >> 6;        // per-row worker id (0..7)
    const int rowg = blockIdx.x * ROWS + r;

    // ---- Phase 1a: encode inputs: m = tanh(x), e = a/(1+a), a = |x|+1e-8 ----
    for (int j = c; j < 35; j += 8) {
        float x; int base;
        if (j < 25)      { x = proprio  [(size_t)rowg * 25 + j];       base = 2 * j; }
        else if (j < 27) { x = tactile  [(size_t)rowg * 2 + (j - 25)]; base = 50 + 2 * (j - 25); }
        else if (j < 33) { x = ee       [(size_t)rowg * 6 + (j - 27)]; base = 54 + 2 * (j - 27); }
        else             { x = internal_[(size_t)rowg * 2 + (j - 33)]; base = 66 + 2 * (j - 33); }
        float a = fabsf(x) + 1e-8f;
        ovl.p1.enc[r][base]     = tanhf(x);
        ovl.p1.enc[r][base + 1] = a / (1.0f + a);   // sigmoid(log(a))
    }
    __syncthreads();

    // ---- Phase 1b: small-MLP layer 1 (fp32) ----
    layer1<8, 32> (ovl.p1.enc[r], FG, f_w1, f_b1, &ovl.p1.h1s[r][0],   c, 8);
    layer1<18,64> (ovl.p1.enc[r], WG, w_w1, w_b1, &ovl.p1.h1s[r][32],  c, 8);
    layer1<24,64> (ovl.p1.enc[r], AG, a_w1, a_b1, &ovl.p1.h1s[r][96],  c, 8);
    layer1<4, 16> (ovl.p1.enc[r], HG, h_w1, h_b1, &ovl.p1.h1s[r][160], c, 8);
    layer1<12,32> (ovl.p1.enc[r], EG, e_w1, e_b1, &ovl.p1.h1s[r][176], c, 8);
    layer1<4, 16> (ovl.p1.enc[r], IG, i_w1, i_b1, &ovl.p1.h1s[r][208], c, 8);
    __syncthreads();

    // ---- Phase 1c: small-MLP layer 2 -> fused (f16) ----
    layer2<32, 64>(&ovl.p1.h1s[r][0],   f_w2, f_b2, &fusedS[r][0],   c, 8);
    layer2<64, 64>(&ovl.p1.h1s[r][32],  w_w2, w_b2, &fusedS[r][64],  c, 8);
    layer2<64,128>(&ovl.p1.h1s[r][96],  a_w2, a_b2, &fusedS[r][128], c, 8);
    layer2<16, 32>(&ovl.p1.h1s[r][160], h_w2, h_b2, &fusedS[r][256], c, 8);
    layer2<32, 64>(&ovl.p1.h1s[r][176], e_w2, e_b2, &fusedS[r][288], c, 8);
    layer2<16, 32>(&ovl.p1.h1s[r][208], i_w2, i_b2, &fusedS[r][352], c, 8);
    __syncthreads();   // phase-1 scratch dead from here; ovl.hbuf may be written

    // ---- Phase 2: WMMA GEMMs. 16 waves: (wv&7) -> 64-col slice, (wv>>3) -> 32-row half.
    // Waves sharing a column slice fetch identical B fragments -> WGP$ hits. ----
    const int lane  = tid & 31;
    const int wv    = tid >> 5;       // 0..15
    const int colw  = wv & 7;         // column-slice id
    const int mbase = (wv >> 3) * 32; // row base for this wave's two M-tiles
    const int lrow  = lane & 15;
    const int hi    = (lane >= 16) ? 1 : 0;
    const int koffA = hi ? 8 : 0;

    // GEMM 1: fused(64x384) @ fu_w1(384x512) -> hidden slice 32x64 per wave
    v8f acc[2][4];
#pragma unroll
    for (int mt = 0; mt < 2; ++mt)
#pragma unroll
        for (int nt = 0; nt < 4; ++nt) acc[mt][nt] = v8f{};

    for (int kb = 0; kb < 12; ++kb) {
        v16h A[2];
#pragma unroll
        for (int mt = 0; mt < 2; ++mt) {
            half8 alo = *(const half8*)&fusedS[mbase + mt * 16 + lrow][kb * 32 + koffA];
            half8 ahi = *(const half8*)&fusedS[mbase + mt * 16 + lrow][kb * 32 + koffA + 16];
#pragma unroll
            for (int i = 0; i < 8; ++i) { A[mt][i] = alo[i]; A[mt][i + 8] = ahi[i]; }
        }
#pragma unroll
        for (int nt = 0; nt < 4; ++nt) {
            const int f = kb * 32 + (colw * 4 + nt);
            const half8* bp = (const half8*)(pW1 + ((size_t)f << 9) + (lane << 4));
            half8 b0 = bp[0], b1 = bp[1];
            v16h Bf;
#pragma unroll
            for (int i = 0; i < 8; ++i) { Bf[i] = b0[i]; Bf[i + 8] = b1[i]; }
            acc[0][nt] = __builtin_amdgcn_wmma_f32_16x16x32_f16(
                false, A[0], false, Bf, (short)0, acc[0][nt], false, false);
            acc[1][nt] = __builtin_amdgcn_wmma_f32_16x16x32_f16(
                false, A[1], false, Bf, (short)0, acc[1][nt], false, false);
        }
    }
    // bias + relu -> hidden LDS (f16); overlays dead phase-1 scratch
#pragma unroll
    for (int nt = 0; nt < 4; ++nt) {
        const int col = colw * 64 + nt * 16 + lrow;
        const float b = fu_b1[col];
#pragma unroll
        for (int mt = 0; mt < 2; ++mt)
#pragma unroll
            for (int rr = 0; rr < 8; ++rr) {
                const int row = mbase + mt * 16 + rr + (hi ? 8 : 0);
                ovl.hbuf[row][col] = (_Float16)fmaxf(acc[mt][nt][rr] + b, 0.0f);
            }
    }
    if (tid < ROWS) { sums[tid] = 0.0f; sums2[tid] = 0.0f; }
    __syncthreads();

    // GEMM 2: hidden(64x512) @ fu_w2(512x512) -> z slice 32x64 per wave
    v8f acc2[2][4];
#pragma unroll
    for (int mt = 0; mt < 2; ++mt)
#pragma unroll
        for (int nt = 0; nt < 4; ++nt) acc2[mt][nt] = v8f{};

    for (int kb = 0; kb < 16; ++kb) {
        v16h A[2];
#pragma unroll
        for (int mt = 0; mt < 2; ++mt) {
            half8 alo = *(const half8*)&ovl.hbuf[mbase + mt * 16 + lrow][kb * 32 + koffA];
            half8 ahi = *(const half8*)&ovl.hbuf[mbase + mt * 16 + lrow][kb * 32 + koffA + 16];
#pragma unroll
            for (int i = 0; i < 8; ++i) { A[mt][i] = alo[i]; A[mt][i + 8] = ahi[i]; }
        }
#pragma unroll
        for (int nt = 0; nt < 4; ++nt) {
            const int f = kb * 32 + (colw * 4 + nt);
            const half8* bp = (const half8*)(pW2 + ((size_t)f << 9) + (lane << 4));
            half8 b0 = bp[0], b1 = bp[1];
            v16h Bf;
#pragma unroll
            for (int i = 0; i < 8; ++i) { Bf[i] = b0[i]; Bf[i + 8] = b1[i]; }
            acc2[0][nt] = __builtin_amdgcn_wmma_f32_16x16x32_f16(
                false, A[0], false, Bf, (short)0, acc2[0][nt], false, false);
            acc2[1][nt] = __builtin_amdgcn_wmma_f32_16x16x32_f16(
                false, A[1], false, Bf, (short)0, acc2[1][nt], false, false);
        }
    }

    float bias2[4];
#pragma unroll
    for (int nt = 0; nt < 4; ++nt) bias2[nt] = fu_b2[colw * 64 + nt * 16 + lrow];

    // per-row partial sums for LayerNorm (ds_add_f32 atomics)
#pragma unroll
    for (int mt = 0; mt < 2; ++mt)
#pragma unroll
        for (int rr = 0; rr < 8; ++rr) {
            float ps = 0.0f, ps2 = 0.0f;
#pragma unroll
            for (int nt = 0; nt < 4; ++nt) {
                float z = acc2[mt][nt][rr] + bias2[nt];
                ps += z; ps2 += z * z;
            }
            const int row = mbase + mt * 16 + rr + (hi ? 8 : 0);
            atomicAdd(&sums[row], ps);
            atomicAdd(&sums2[row], ps2);
        }
    __syncthreads();

    // ---- normalize + affine + store ----
    float g[4], bb[4];
#pragma unroll
    for (int nt = 0; nt < 4; ++nt) {
        const int col = colw * 64 + nt * 16 + lrow;
        g[nt] = ln_g[col]; bb[nt] = ln_b[col];
    }
#pragma unroll
    for (int mt = 0; mt < 2; ++mt)
#pragma unroll
        for (int rr = 0; rr < 8; ++rr) {
            const int row = mbase + mt * 16 + rr + (hi ? 8 : 0);
            const float mu  = sums[row]  * (1.0f / 512.0f);
            const float var = sums2[row] * (1.0f / 512.0f) - mu * mu;
            const float inv = rsqrtf(var + 1e-5f);
            const size_t obase = ((size_t)(blockIdx.x * ROWS + row)) * 512;
#pragma unroll
            for (int nt = 0; nt < 4; ++nt) {
                const int col = colw * 64 + nt * 16 + lrow;
                float z = acc2[mt][nt][rr] + bias2[nt];
                out[obase + col] = (z - mu) * inv * g[nt] + bb[nt];
            }
        }
}

extern "C" void kernel_launch(void* const* d_in, const int* in_sizes, int n_in,
                              void* d_out, int out_size, void* d_ws, size_t ws_size,
                              hipStream_t stream) {
    const float* proprio  = (const float*)d_in[0];
    const float* tactile  = (const float*)d_in[1];
    const float* ee       = (const float*)d_in[2];
    const float* internal_= (const float*)d_in[3];
    const float* f_w1 = (const float*)d_in[4];  const float* f_b1 = (const float*)d_in[5];
    const float* f_w2 = (const float*)d_in[6];  const float* f_b2 = (const float*)d_in[7];
    const float* w_w1 = (const float*)d_in[8];  const float* w_b1 = (const float*)d_in[9];
    const float* w_w2 = (const float*)d_in[10]; const float* w_b2 = (const float*)d_in[11];
    const float* a_w1 = (const float*)d_in[12]; const float* a_b1 = (const float*)d_in[13];
    const float* a_w2 = (const float*)d_in[14]; const float* a_b2 = (const float*)d_in[15];
    const float* h_w1 = (const float*)d_in[16]; const float* h_b1 = (const float*)d_in[17];
    const float* h_w2 = (const float*)d_in[18]; const float* h_b2 = (const float*)d_in[19];
    const float* e_w1 = (const float*)d_in[20]; const float* e_b1 = (const float*)d_in[21];
    const float* e_w2 = (const float*)d_in[22]; const float* e_b2 = (const float*)d_in[23];
    const float* i_w1 = (const float*)d_in[24]; const float* i_b1 = (const float*)d_in[25];
    const float* i_w2 = (const float*)d_in[26]; const float* i_b2 = (const float*)d_in[27];
    const float* fu_w1 = (const float*)d_in[28]; const float* fu_b1 = (const float*)d_in[29];
    const float* fu_w2 = (const float*)d_in[30]; const float* fu_b2 = (const float*)d_in[31];
    const float* ln_g  = (const float*)d_in[32]; const float* ln_b  = (const float*)d_in[33];
    float* outp = (float*)d_out;

    const int B = in_sizes[0] / 25;          // 131072

    _Float16* pW1 = (_Float16*)d_ws;                       // 384*512 f16 = 384 KB
    _Float16* pW2 = pW1 + (size_t)384 * 512;               // 512*512 f16 = 512 KB

    // pack fp32 weights into WMMA-B fragment layout (f16), L2-resident
    {
        int total1 = 384 * 512;
        pack_wmma_b<<<(total1 + 255) / 256, 256, 0, stream>>>(fu_w1, pW1, 512, total1);
        int total2 = 512 * 512;
        pack_wmma_b<<<(total2 + 255) / 256, 256, 0, stream>>>(fu_w2, pW2, 512, total2);
    }

    proprio_fused_kernel<<<B / ROWS, BLK, 0, stream>>>(
        proprio, tactile, ee, internal_,
        f_w1, f_b1, f_w2, f_b2,
        w_w1, w_b1, w_w2, w_b2,
        a_w1, a_b1, a_w2, a_b2,
        h_w1, h_b1, h_w2, h_b2,
        e_w1, e_b1, e_w2, e_b2,
        i_w1, i_b1, i_w2, i_b2,
        pW1, fu_b1, pW2, fu_b2,
        ln_g, ln_b, outp);
}
